// MyRobustGATModel_12180527252133
// MI455X (gfx1250) — matrix operand order repaired and checked by reading, not compiled, not run on hardware
//
#include <hip/hip_runtime.h>

typedef __attribute__((ext_vector_type(16))) _Float16 v16h;
typedef __attribute__((ext_vector_type(8)))  float    v8f;

#define NEGF (-1e9f)

// ---------------------------------------------------------------------------
// CAS-based float atomic max (no NaNs in this workload)
// ---------------------------------------------------------------------------
__device__ inline void atomicMaxF(float* addr, float val) {
    int* ai = (int*)addr;
    int old = __float_as_int(*addr);
    while (__int_as_float(old) < val) {
        int assumed = old;
        old = atomicCAS(ai, assumed, __float_as_int(val));
        if (old == assumed) break;
    }
}

// ---------------------------------------------------------------------------
// fill
// ---------------------------------------------------------------------------
__global__ void fill_kernel(float* p, float v, long long n) {
    long long i = (long long)blockIdx.x * blockDim.x + threadIdx.x;
    long long stride = (long long)gridDim.x * blockDim.x;
    for (; i < n; i += stride) p[i] = v;
}

// ---------------------------------------------------------------------------
// node encoder: x[n, :] = node_emb[x_nodes[n], :]
// ---------------------------------------------------------------------------
__global__ void node_encode_kernel(const int* xn, const float* emb, float* x, int N) {
    long long tid = (long long)blockIdx.x * blockDim.x + threadIdx.x;
    if (tid >= (long long)N * 128) return;
    int n = (int)(tid >> 7), ch = (int)(tid & 127);
    x[(size_t)n * 128 + ch] = emb[(size_t)xn[n] * 128 + ch];
}

// ---------------------------------------------------------------------------
// WMMA GEMM, K as template constant (128 / 64 / 8).
// Y[M,Ncols] = A[M,K](f32) * B[K,Ncols](f32) (+ bias), f16 in / f32 accumulate
// via v_wmma_f32_16x16x32_f16. Block = 256 threads = 8 waves; each wave owns a
// 16x16 output tile; the 16-column B tile is staged once in LDS (f16,
// transposed to [n][K], zero-padded to K=32 for the K=8 edge encoder).
// Fragment packing uses contiguous k = 16*hi + s for A and B: a lane's A elements are one
// contiguous 16-float run -> four global_load_b128; B fragments are two
// ds_load_b128 from the padded LDS tile.
// ---------------------------------------------------------------------------
template <int K>
__global__ void gemm_wmma_kernel(const float* __restrict__ A,
                                 const float* __restrict__ B,
                                 const float* __restrict__ bias,
                                 float* __restrict__ Y,
                                 int M, int Ncols) {
    constexpr int KP = (K < 32) ? 32 : K;       // padded K (zero-filled)
    constexpr int LDB = KP + 16;                // halves; keeps 32B alignment
    __shared__ _Float16 Bs[16][LDB];

    int wave = threadIdx.x >> 5;
    int lane = threadIdx.x & 31;
    int tileM = (blockIdx.x * 8 + wave) * 16;
    int tileN = blockIdx.y * 16;

    // cooperative B tile stage: Bs[n][kk] = (f16) B[kk, tileN + n]
    for (int i = threadIdx.x; i < 16 * KP; i += 256) {
        int n  = i / KP;
        int kk = i % KP;
        float v = (kk < K) ? B[(size_t)kk * Ncols + tileN + n] : 0.0f;
        Bs[n][kk] = (_Float16)v;
    }
    __syncthreads();                             // all waves reach the barrier

    if (tileM >= M) return;                      // wave-uniform exit after sync

    int mrow = lane & 15;
    int hi   = lane >> 4;                        // 0 or 1 (lane half)
    int arow = tileM + mrow; if (arow > M - 1) arow = M - 1;
    const float* Arow = A + (size_t)arow * K;

    v8f acc = {};
    #pragma unroll
    for (int k0 = 0; k0 < KP; k0 += 32) {
        // ---- A fragment ----
        v16h a;
        if constexpr (K >= 32) {
            const float* ab = Arow + k0 + hi * 16;
            if (k0 + 32 < K) __builtin_prefetch(ab + 32, 0, 1);
            float4 q0 = *(const float4*)(ab);
            float4 q1 = *(const float4*)(ab + 4);
            float4 q2 = *(const float4*)(ab + 8);
            float4 q3 = *(const float4*)(ab + 12);
            a[0]=(_Float16)q0.x;  a[1]=(_Float16)q0.y;  a[2]=(_Float16)q0.z;  a[3]=(_Float16)q0.w;
            a[4]=(_Float16)q1.x;  a[5]=(_Float16)q1.y;  a[6]=(_Float16)q1.z;  a[7]=(_Float16)q1.w;
            a[8]=(_Float16)q2.x;  a[9]=(_Float16)q2.y;  a[10]=(_Float16)q2.z; a[11]=(_Float16)q2.w;
            a[12]=(_Float16)q3.x; a[13]=(_Float16)q3.y; a[14]=(_Float16)q3.z; a[15]=(_Float16)q3.w;
        } else {                                 // K == 8: only lanes 0-15, kk<8
            #pragma unroll
            for (int j = 0; j < 16; ++j) a[j] = (_Float16)0.0f;
            if (hi == 0) {
                float4 q0 = *(const float4*)(Arow);
                float4 q1 = *(const float4*)(Arow + 4);
                a[0]=(_Float16)q0.x; a[1]=(_Float16)q0.y; a[2]=(_Float16)q0.z; a[3]=(_Float16)q0.w;
                a[4]=(_Float16)q1.x; a[5]=(_Float16)q1.y; a[6]=(_Float16)q1.z; a[7]=(_Float16)q1.w;
            }
        }
        // ---- B fragment: 16 contiguous halves from LDS (32B aligned) ----
        v16h b = *(const v16h*)(&Bs[mrow][k0 + (hi ? 16 : 0)]);

        acc = __builtin_amdgcn_wmma_f32_16x16x32_f16(
                  false, a, false, b, (short)0, acc, false, false);
    }

    // C/D layout: element r -> row = tileM + r + 8*hi, col = tileN + (lane&15)
    #pragma unroll
    for (int r = 0; r < 8; ++r) {
        int row = tileM + r + (hi ? 8 : 0);
        int col = tileN + mrow;
        if (row < M) {
            float v = acc[r] + (bias ? bias[col] : 0.0f);
            Y[(size_t)row * Ncols + col] = v;
        }
    }
}

// ---------------------------------------------------------------------------
// self-loop attr: masked count + masked sum + mean
// ---------------------------------------------------------------------------
__global__ void loop_cnt_kernel(const int* es, const int* ed, float* cnt, int E) {
    int e = blockIdx.x * blockDim.x + threadIdx.x;
    if (e >= E) return;
    if (es[e] != ed[e]) atomicAdd(&cnt[ed[e]], 1.0f);
}

__global__ void loop_acc_kernel(const int* es, const int* ed, const float* ea,
                                float* lacc, int E) {
    long long tid = (long long)blockIdx.x * blockDim.x + threadIdx.x;
    if (tid >= (long long)E * 64) return;
    int e = (int)(tid >> 6), f = (int)(tid & 63);
    if (es[e] != ed[e])
        atomicAdd(&lacc[(size_t)ed[e] * 64 + f], ea[(size_t)e * 64 + f]);
}

__global__ void loop_fin_kernel(const float* lacc, const float* cnt,
                                float* ea, int N, int E) {
    long long tid = (long long)blockIdx.x * blockDim.x + threadIdx.x;
    if (tid >= (long long)N * 64) return;
    int n = (int)(tid >> 6), f = (int)(tid & 63);
    ea[(size_t)(E + n) * 64 + f] = lacc[(size_t)n * 64 + f] / fmaxf(cnt[n], 1.0f);
}

// ---------------------------------------------------------------------------
// attention logits (per edge, per head) over one ep chunk + segment max
// ---------------------------------------------------------------------------
__global__ void attn_logits_kernel(const int* es, const int* ed,
                                   const float* xl, const float* xr,
                                   const float* epc, const float* att,
                                   float* logits, float* mbuf,
                                   int chunkStart, int chunkLen, int E) {
    long long tid = (long long)blockIdx.x * blockDim.x + threadIdx.x;
    if (tid >= (long long)chunkLen * 8) return;
    int le = (int)(tid >> 3), h = (int)(tid & 7);
    int e = chunkStart + le;
    int s, d; bool valid;
    if (e < E) { s = es[e]; d = ed[e]; valid = (s != d); }
    else       { s = d = e - E;        valid = true; }
    float logit = NEGF;
    if (valid) {
        const float4* pl = (const float4*)(xl  + (size_t)s  * 128 + h * 16);
        const float4* pr = (const float4*)(xr  + (size_t)d  * 128 + h * 16);
        const float4* pe = (const float4*)(epc + (size_t)le * 128 + h * 16);
        const float4* pa = (const float4*)(att + h * 16);
        float accv = 0.0f;
        #pragma unroll
        for (int q = 0; q < 4; ++q) {
            float4 vl = pl[q], vr = pr[q], ve = pe[q], va = pa[q];
            float z0 = vl.x + vr.x + ve.x;  z0 = (z0 > 0.f) ? z0 : 0.2f * z0;
            float z1 = vl.y + vr.y + ve.y;  z1 = (z1 > 0.f) ? z1 : 0.2f * z1;
            float z2 = vl.z + vr.z + ve.z;  z2 = (z2 > 0.f) ? z2 : 0.2f * z2;
            float z3 = vl.w + vr.w + ve.w;  z3 = (z3 > 0.f) ? z3 : 0.2f * z3;
            accv += z0 * va.x + z1 * va.y + z2 * va.z + z3 * va.w;
        }
        logit = accv;
        atomicMaxF(&mbuf[d * 8 + h], logit);
    }
    logits[(size_t)e * 8 + h] = logit;
}

// ---------------------------------------------------------------------------
// p = exp(logit - m[d]) (masked), denominator accumulation. p overwrites logits.
// ---------------------------------------------------------------------------
__global__ void attn_exp_kernel(const int* ed, const float* mbuf,
                                float* logits, float* denom, int E, int ES) {
    long long tid = (long long)blockIdx.x * blockDim.x + threadIdx.x;
    if (tid >= (long long)ES * 8) return;
    int e = (int)(tid >> 3), h = (int)(tid & 7);
    int d = (e < E) ? ed[e] : (e - E);
    float lg = logits[tid];
    float p = 0.0f;
    if (lg > 0.5f * NEGF) p = __expf(lg - mbuf[d * 8 + h]);
    logits[tid] = p;
    atomicAdd(&denom[d * 8 + h], p);
}

// ---------------------------------------------------------------------------
// h[d, :] += alpha * xl[s, :]
// ---------------------------------------------------------------------------
__global__ void aggregate_kernel(const int* es, const int* ed,
                                 const float* pbuf, const float* denom,
                                 const float* xl, float* hacc, int E, int ES) {
    long long tid = (long long)blockIdx.x * blockDim.x + threadIdx.x;
    if (tid >= (long long)ES * 128) return;
    int e = (int)(tid >> 7), ch = (int)(tid & 127);
    int h = ch >> 4;
    int s, d;
    if (e < E) { s = es[e]; d = ed[e]; }
    else       { s = d = e - E; }
    float p = pbuf[(size_t)e * 8 + h];
    if (p != 0.0f) {
        float alpha = p / denom[d * 8 + h];
        atomicAdd(&hacc[(size_t)d * 128 + ch], alpha * xl[(size_t)s * 128 + ch]);
    }
}

// ---------------------------------------------------------------------------
// node update: +gbias, eval-mode BN (mean=0,var=1), ELU
// ---------------------------------------------------------------------------
__global__ void node_update_kernel(const float* hacc, const float* gbias,
                                   const float* bng, const float* bnb,
                                   float* x, int N) {
    long long tid = (long long)blockIdx.x * blockDim.x + threadIdx.x;
    if (tid >= (long long)N * 128) return;
    int ch = (int)(tid & 127);
    float inv = rsqrtf(1.0f + 1e-5f);
    float h = hacc[tid] + gbias[ch];
    h = bng[ch] * h * inv + bnb[ch];
    x[tid] = (h > 0.0f) ? h : (__expf(h) - 1.0f);
}

// ---------------------------------------------------------------------------
// global mean pool
// ---------------------------------------------------------------------------
__global__ void pool_sum_kernel(const int* batch, const float* x, float* gsum, int N) {
    long long tid = (long long)blockIdx.x * blockDim.x + threadIdx.x;
    if (tid >= (long long)N * 128) return;
    int n = (int)(tid >> 7), ch = (int)(tid & 127);
    atomicAdd(&gsum[(size_t)batch[n] * 128 + ch], x[tid]);
}

__global__ void pool_cnt_kernel(const int* batch, float* gcnt, int N) {
    int n = blockIdx.x * blockDim.x + threadIdx.x;
    if (n >= N) return;
    atomicAdd(&gcnt[batch[n]], 1.0f);
}

// ---------------------------------------------------------------------------
// head: out[g, c] = (gsum[g]/cnt[g]) . head_W[:, c] + head_b[c]
// ---------------------------------------------------------------------------
__global__ void head_kernel(const float* gsum, const float* gcnt,
                            const float* hW, const float* hb,
                            float* out, int G, int NCLS) {
    int tid = blockIdx.x * blockDim.x + threadIdx.x;
    if (tid >= G * NCLS) return;
    int g = tid / NCLS, c = tid % NCLS;
    float invc = 1.0f / fmaxf(gcnt[g], 1.0f);
    float accv = 0.0f;
    for (int k = 0; k < 128; ++k)
        accv += gsum[(size_t)g * 128 + k] * invc * hW[(size_t)k * NCLS + c];
    out[tid] = accv + hb[c];
}

// ===========================================================================
extern "C" void kernel_launch(void* const* d_in, const int* in_sizes, int n_in,
                              void* d_out, int out_size, void* d_ws, size_t ws_size,
                              hipStream_t stream) {
    const int N = 50000, E = 800000, G = 512, D = 128, HH = 8, EE = 64,
              L = 4, NCLS = 10;
    const int ES = E + N;
    const int CH = 106256;           // ep chunk rows, multiple of 16

    const int*   x_nodes   = (const int*)  d_in[0];
    const int*   esrc      = (const int*)  d_in[1];
    const int*   edst      = (const int*)  d_in[2];
    const float* edge_attr = (const float*)d_in[3];
    const int*   batch     = (const int*)  d_in[4];
    const float* node_emb  = (const float*)d_in[5];
    const float* edge_W    = (const float*)d_in[6];
    const float* edge_b    = (const float*)d_in[7];
    const float* Wl        = (const float*)d_in[8];
    const float* bl        = (const float*)d_in[9];
    const float* Wr        = (const float*)d_in[10];
    const float* br        = (const float*)d_in[11];
    const float* We        = (const float*)d_in[12];
    const float* att       = (const float*)d_in[13];
    const float* gbias     = (const float*)d_in[14];
    const float* bng       = (const float*)d_in[15];
    const float* bnb       = (const float*)d_in[16];
    const float* headW     = (const float*)d_in[17];
    const float* headb     = (const float*)d_in[18];
    float* out = (float*)d_out;
    (void)in_sizes; (void)n_in; (void)out_size; (void)ws_size;

    // workspace partition (floats)
    float* w = (float*)d_ws;
    float* x    = w; w += (size_t)N * D;
    float* xl   = w; w += (size_t)N * D;
    float* xr   = w; w += (size_t)N * D;
    float* hacc = w; w += (size_t)N * D;
    float* ea   = w; w += (size_t)ES * EE;
    float* epc  = w; w += (size_t)CH * D;
    float* pbuf = w; w += (size_t)ES * HH;     // logits, then p
    float* mbuf = w; w += (size_t)N * HH;
    float* den  = w; w += (size_t)N * HH;
    float* lacc = w; w += (size_t)N * EE;
    float* cntb = w; w += (size_t)N;
    float* gsum = w; w += (size_t)G * D;
    float* gcnt = w; w += (size_t)G;

    auto nblk = [](long long n, int b) { return (unsigned)((n + b - 1) / b); };
    auto gemm = [&](const float* A, const float* B, const float* bias, float* Y,
                    int M, int K, int Ncols) {
        int mtiles = (M + 15) / 16;
        dim3 grid((unsigned)((mtiles + 7) / 8), (unsigned)(Ncols / 16));
        if (K == 128)
            gemm_wmma_kernel<128><<<grid, 256, 0, stream>>>(A, B, bias, Y, M, Ncols);
        else if (K == 64)
            gemm_wmma_kernel<64><<<grid, 256, 0, stream>>>(A, B, bias, Y, M, Ncols);
        else
            gemm_wmma_kernel<8><<<grid, 256, 0, stream>>>(A, B, bias, Y, M, Ncols);
    };

    // ---- setup ----
    fill_kernel<<<2048, 256, 0, stream>>>(lacc, 0.0f, (long long)N * EE);
    fill_kernel<<<2048, 256, 0, stream>>>(cntb, 0.0f, (long long)N);
    fill_kernel<<<2048, 256, 0, stream>>>(gsum, 0.0f, (long long)G * D);
    fill_kernel<<<64,   256, 0, stream>>>(gcnt, 0.0f, (long long)G);

    node_encode_kernel<<<nblk((long long)N * D, 256), 256, 0, stream>>>(
        x_nodes, node_emb, x, N);

    // edge encoder: eattr = edge_attr @ edge_W + edge_b  (K=8, zero-padded)
    gemm(edge_attr, edge_W, edge_b, ea, E, 8, EE);

    // self-loop attrs (PyG fill_value='mean')
    loop_cnt_kernel<<<nblk(E, 256), 256, 0, stream>>>(esrc, edst, cntb, E);
    loop_acc_kernel<<<nblk((long long)E * EE, 256), 256, 0, stream>>>(
        esrc, edst, ea, lacc, E);
    loop_fin_kernel<<<nblk((long long)N * EE, 256), 256, 0, stream>>>(
        lacc, cntb, ea, N, E);

    // ---- GAT layers ----
    for (int i = 0; i < L; ++i) {
        gemm(x, Wl + (size_t)i * D * D, bl + (size_t)i * D, xl, N, D, D);
        gemm(x, Wr + (size_t)i * D * D, br + (size_t)i * D, xr, N, D, D);

        fill_kernel<<<512,  256, 0, stream>>>(mbuf, NEGF, (long long)N * HH);
        fill_kernel<<<512,  256, 0, stream>>>(den,  0.0f, (long long)N * HH);
        fill_kernel<<<2048, 256, 0, stream>>>(hacc, 0.0f, (long long)N * D);

        // chunked ep = ea @ We[i], fused with logits/segment-max
        for (int start = 0; start < ES; start += CH) {
            int len = (ES - start < CH) ? (ES - start) : CH;
            gemm(ea + (size_t)start * EE, We + (size_t)i * EE * D, nullptr,
                 epc, len, EE, D);
            attn_logits_kernel<<<nblk((long long)len * HH, 256), 256, 0, stream>>>(
                esrc, edst, xl, xr, epc, att + (size_t)i * D,
                pbuf, mbuf, start, len, E);
        }

        attn_exp_kernel<<<nblk((long long)ES * HH, 256), 256, 0, stream>>>(
            edst, mbuf, pbuf, den, E, ES);
        aggregate_kernel<<<nblk((long long)ES * D, 256), 256, 0, stream>>>(
            esrc, edst, pbuf, den, xl, hacc, E, ES);
        node_update_kernel<<<nblk((long long)N * D, 256), 256, 0, stream>>>(
            hacc, gbias + (size_t)i * D, bng + (size_t)i * D,
            bnb + (size_t)i * D, x, N);
    }

    // ---- pool + head ----
    pool_sum_kernel<<<nblk((long long)N * D, 256), 256, 0, stream>>>(batch, x, gsum, N);
    pool_cnt_kernel<<<nblk(N, 256), 256, 0, stream>>>(batch, gcnt, N);
    head_kernel<<<nblk(G * NCLS, 256), 256, 0, stream>>>(
        gsum, gcnt, headW, headb, out, G, NCLS);
}
